// MLP_4947802325695
// MI455X (gfx1250) — compile-verified, matrix operand order checked by reading, not run on hardware
//
#include <hip/hip_runtime.h>
#include <stdint.h>

// ---------------------------------------------------------------------------
// NF4 SwiGLU MLP for MI455X (gfx1250, wave32, WMMA).
//   H = silu(x @ Wg^T) * (x @ Wu^T)   -> bf16 workspace [T, m]
//   out = H @ Wd^T                    -> f32 [T, hd]
//
// LDS tiles staged in the per-lane WMMA fragment layout ([subtile][lane][16
// halfs], 32B contiguous per lane) -> each operand fragment is one 32-byte
// LDS read (2x ds_load_b128).  Register software pipeline overlaps next-tile
// global loads with WMMA.  Each wave computes a 32x32 sub-tile (2x2 WMMA
// tiles) so every LDS fragment is reused twice: 12 b128 loads per 8 WMMAs.
// ---------------------------------------------------------------------------

typedef __attribute__((ext_vector_type(16))) __bf16 v16bf;
typedef __attribute__((ext_vector_type(8)))  float  v8f;

#define HD   4096
#define MDIM 14336
#define TDIM 2048

__device__ __constant__ float NF4_TAB[16] = {
    -1.0f, -0.6961928009986877f, -0.5250730514526367f, -0.39491748809814453f,
    -0.28444138169288635f, -0.18477343022823334f, -0.09105003625154495f, 0.0f,
    0.07958029955625534f, 0.16093020141124725f, 0.24611230194568634f,
    0.33791524171829224f, 0.44070982933044434f, 0.5626170039176941f,
    0.7229568362236023f, 1.0f };

// float -> bf16 bits (RNE), result in low 16 bits
__device__ inline unsigned int f2bf(float f) {
  unsigned int u = __builtin_bit_cast(unsigned int, f);
  u += 0x7FFFu + ((u >> 16) & 1u);
  return u >> 16;
}

// pack two f32 -> packed bf16 pair (one v_cvt_pk_bf16_f32 when available)
__device__ inline unsigned int pkbf(float a, float b) {
#if __has_builtin(__builtin_amdgcn_cvt_pk_bf16_f32)
  auto p = __builtin_amdgcn_cvt_pk_bf16_f32(a, b);
  return __builtin_bit_cast(unsigned int, p);
#else
  return f2bf(a) | (f2bf(b) << 16);
#endif
}

// NF4 codebook lookup: codes are in [0,16) so no masking needed; gather from
// NF4_TAB[lane&15] held in each lane (ds_bpermute_b32, byte index = c*4).
__device__ inline float lut(int c, float nf4lane) {
  int r = __builtin_amdgcn_ds_bpermute(c << 2,
                                       __builtin_bit_cast(int, nf4lane));
  return __builtin_bit_cast(float, r);
}

// Dequantize 8 NF4 codes (two int4s) -> 4 packed bf16 pairs.
__device__ inline uint4 dq8(int4 c0, int4 c1, float am, float nf4lane) {
  uint4 r;
  r.x = pkbf(lut(c0.x, nf4lane) * am, lut(c0.y, nf4lane) * am);
  r.y = pkbf(lut(c0.z, nf4lane) * am, lut(c0.w, nf4lane) * am);
  r.z = pkbf(lut(c1.x, nf4lane) * am, lut(c1.y, nf4lane) * am);
  r.w = pkbf(lut(c1.z, nf4lane) * am, lut(c1.w, nf4lane) * am);
  return r;
}

// Fragment-layout note (ISA 7.12.2, 16-bit 16x32 operand):
//   lane = (row & 15) + 16*hi ; lane's 16 elements j=0..15 map to
//   k = 8*hi + (j < 8 ? j : j + 8)  i.e. chunks {0..7} and {16..23} (+8*hi).
// A writer thread owning (row, hi) therefore loads 4 contiguous 4-element
// chunks at k = 8*hi + {0,4,16,20} and writes 32 contiguous LDS bytes.

// ---------------------------------------------------------------------------
// Kernel 1: fused gate/up.  Tile 64(T) x 128(m) per 256-thread workgroup.
// 8 waves: waveM in [0,2) x waveN in [0,4); each wave owns 32x32 of G and U.
// ---------------------------------------------------------------------------
__global__ __launch_bounds__(256) void gateup_kernel(
    const float* __restrict__ x,
    const int*   __restrict__ gcodes, const float* __restrict__ gamax,
    const int*   __restrict__ ucodes, const float* __restrict__ uamax,
    unsigned short* __restrict__ H) {
  // fragment-layout LDS: [subtile][lane][16]
  __shared__ __align__(32) unsigned short sA[4 * 32 * 16];   // 64 rows of A
  __shared__ __align__(32) unsigned short sG[8 * 32 * 16];   // 128 rows Wg
  __shared__ __align__(32) unsigned short sU[8 * 32 * 16];   // 128 rows Wu

  const int tid   = threadIdx.x;
  const int lane  = tid & 31;
  const int wave  = tid >> 5;
  const int waveM = wave >> 2;            // 0..1 : 32 rows
  const int waveN = wave & 3;             // 0..3 : 32 cols
  const int nbase = blockIdx.x * 128;     // m tile
  const int tbase = blockIdx.y * 64;      // T tile

  const float nf4lane = NF4_TAB[lane & 15];

  // ---- writer roles -------------------------------------------------------
  const int wn  = tid >> 1;               // 0..127
  const int whi = tid & 1;
  unsigned short* wgdst = sG + (((wn >> 4) * 32) + (wn & 15) + 16 * whi) * 16;
  unsigned short* wudst = sU + (((wn >> 4) * 32) + (wn & 15) + 16 * whi) * 16;
  const int*   gb  = gcodes + (size_t)(nbase + wn) * HD + 8 * whi;
  const int*   ub  = ucodes + (size_t)(nbase + wn) * HD + 8 * whi;
  const float* gab = gamax + (size_t)(nbase + wn) * (HD / 64);
  const float* uab = uamax + (size_t)(nbase + wn) * (HD / 64);

  const bool aact = tid < 128;
  const int  arow = (tid >> 1) & 63;
  const int  ahi  = tid & 1;
  const float* ab = x + (size_t)(tbase + arow) * HD + 8 * ahi;
  unsigned short* adst =
      sA + (((arow >> 4) * 32) + (arow & 15) + 16 * ahi) * 16;

  // ---- pipeline registers -------------------------------------------------
  int4  gc[4], uc[4];
  float ga, ua;
  float4 af[4];

  auto fetch_tiles = [&](int kb) {
    const int* gp = gb + kb;
    gc[0] = *(const int4*)(gp);      gc[1] = *(const int4*)(gp + 4);
    gc[2] = *(const int4*)(gp + 16); gc[3] = *(const int4*)(gp + 20);
    const int* up = ub + kb;
    uc[0] = *(const int4*)(up);      uc[1] = *(const int4*)(up + 4);
    uc[2] = *(const int4*)(up + 16); uc[3] = *(const int4*)(up + 20);
    ga = gab[kb >> 6];               // 32-wide k tile sits in one 64-block
    ua = uab[kb >> 6];
    if (aact) {
      const float* ap = ab + kb;
      af[0] = *(const float4*)(ap);      af[1] = *(const float4*)(ap + 4);
      af[2] = *(const float4*)(ap + 16); af[3] = *(const float4*)(ap + 20);
    }
  };

  auto stage_lds = [&]() {
    ((uint4*)wgdst)[0] = dq8(gc[0], gc[1], ga, nf4lane);
    ((uint4*)wgdst)[1] = dq8(gc[2], gc[3], ga, nf4lane);
    ((uint4*)wudst)[0] = dq8(uc[0], uc[1], ua, nf4lane);
    ((uint4*)wudst)[1] = dq8(uc[2], uc[3], ua, nf4lane);
    if (aact) {
      uint4 a0, a1;
      a0.x = pkbf(af[0].x, af[0].y); a0.y = pkbf(af[0].z, af[0].w);
      a0.z = pkbf(af[1].x, af[1].y); a0.w = pkbf(af[1].z, af[1].w);
      a1.x = pkbf(af[2].x, af[2].y); a1.y = pkbf(af[2].z, af[2].w);
      a1.z = pkbf(af[3].x, af[3].y); a1.w = pkbf(af[3].z, af[3].w);
      ((uint4*)adst)[0] = a0;
      ((uint4*)adst)[1] = a1;
    }
  };

  v8f accG[2][2], accU[2][2];
#pragma unroll
  for (int mi = 0; mi < 2; ++mi)
#pragma unroll
    for (int ni = 0; ni < 2; ++ni) { accG[mi][ni] = {}; accU[mi][ni] = {}; }

  const unsigned short* afp = sA + (waveM * 2 * 32 + lane) * 16;
  const unsigned short* gfp = sG + (waveN * 2 * 32 + lane) * 16;
  const unsigned short* ufp = sU + (waveN * 2 * 32 + lane) * 16;

  fetch_tiles(0);
  for (int kb = 0; kb < HD; kb += 32) {
    stage_lds();
    __syncthreads();
    if (kb + 32 < HD) fetch_tiles(kb + 32);   // overlap with WMMA below

    v16bf afr[2], gfr[2], ufr[2];
#pragma unroll
    for (int i = 0; i < 2; ++i) {
      afr[i] = *(const v16bf*)(afp + i * 512);   // 512 halfs = one subtile
      gfr[i] = *(const v16bf*)(gfp + i * 512);
      ufr[i] = *(const v16bf*)(ufp + i * 512);
    }
#pragma unroll
    for (int mi = 0; mi < 2; ++mi)
#pragma unroll
      for (int ni = 0; ni < 2; ++ni) {
        accG[mi][ni] = __builtin_amdgcn_wmma_f32_16x16x32_bf16(
            false, afr[mi], false, gfr[ni], (short)0, accG[mi][ni],
            false, false);
        accU[mi][ni] = __builtin_amdgcn_wmma_f32_16x16x32_bf16(
            false, afr[mi], false, ufr[ni], (short)0, accU[mi][ni],
            false, false);
      }
    __syncthreads();
  }

  // Epilogue: h = silu(g) * u, store bf16.  C layout: VGPR r -> M = r (+8 for
  // lanes 16-31), N = lane % 16.
  const int hi = lane >> 4;
#pragma unroll
  for (int mi = 0; mi < 2; ++mi) {
    const int rowb = tbase + (waveM * 2 + mi) * 16 + hi * 8;
#pragma unroll
    for (int ni = 0; ni < 2; ++ni) {
      const int col = nbase + (waveN * 2 + ni) * 16 + (lane & 15);
#pragma unroll
      for (int r = 0; r < 8; ++r) {
        float g = accG[mi][ni][r];
        float u = accU[mi][ni][r];
        float h = (g / (1.0f + __expf(-g))) * u;
        H[(size_t)(rowb + r) * MDIM + col] = (unsigned short)f2bf(h);
      }
    }
  }
}

// ---------------------------------------------------------------------------
// Kernel 2: down projection.  Tile 64(T) x 128(hd), K loop over m.
// ---------------------------------------------------------------------------
__global__ __launch_bounds__(256) void down_kernel(
    const unsigned short* __restrict__ H,
    const int*   __restrict__ dcodes, const float* __restrict__ damax,
    float* __restrict__ out) {
  __shared__ __align__(32) unsigned short sA[4 * 32 * 16];
  __shared__ __align__(32) unsigned short sW[8 * 32 * 16];

  const int tid   = threadIdx.x;
  const int lane  = tid & 31;
  const int wave  = tid >> 5;
  const int waveM = wave >> 2;            // 0..1
  const int waveN = wave & 3;             // 0..3
  const int nbase = blockIdx.x * 128;     // hd tile
  const int tbase = blockIdx.y * 64;      // T tile

  const float nf4lane = NF4_TAB[lane & 15];

  const int wn  = tid >> 1;
  const int whi = tid & 1;
  unsigned short* wddst = sW + (((wn >> 4) * 32) + (wn & 15) + 16 * whi) * 16;
  const int*   db  = dcodes + (size_t)(nbase + wn) * MDIM + 8 * whi;
  const float* dab = damax + (size_t)(nbase + wn) * (MDIM / 64);

  const bool aact = tid < 128;
  const int  arow = (tid >> 1) & 63;
  const int  ahi  = tid & 1;
  const unsigned short* hb = H + (size_t)(tbase + arow) * MDIM + 8 * ahi;
  unsigned short* adst =
      sA + (((arow >> 4) * 32) + (arow & 15) + 16 * ahi) * 16;

  int4  dc[4];
  float da;
  uint4 ah0, ah1;

  auto fetch_tiles = [&](int kb) {
    const int* dp = db + kb;
    dc[0] = *(const int4*)(dp);      dc[1] = *(const int4*)(dp + 4);
    dc[2] = *(const int4*)(dp + 16); dc[3] = *(const int4*)(dp + 20);
    da = dab[kb >> 6];
    if (aact) {
      ah0 = *(const uint4*)(hb + kb);        // 8 bf16 (16B aligned)
      ah1 = *(const uint4*)(hb + kb + 16);   // 8 bf16
    }
  };

  auto stage_lds = [&]() {
    ((uint4*)wddst)[0] = dq8(dc[0], dc[1], da, nf4lane);
    ((uint4*)wddst)[1] = dq8(dc[2], dc[3], da, nf4lane);
    if (aact) {
      ((uint4*)adst)[0] = ah0;
      ((uint4*)adst)[1] = ah1;
    }
  };

  v8f acc[2][2];
#pragma unroll
  for (int mi = 0; mi < 2; ++mi)
#pragma unroll
    for (int ni = 0; ni < 2; ++ni) acc[mi][ni] = {};

  const unsigned short* afp = sA + (waveM * 2 * 32 + lane) * 16;
  const unsigned short* wfp = sW + (waveN * 2 * 32 + lane) * 16;

  fetch_tiles(0);
  for (int kb = 0; kb < MDIM; kb += 32) {
    stage_lds();
    __syncthreads();
    if (kb + 32 < MDIM) fetch_tiles(kb + 32);

    v16bf afr[2], wfr[2];
#pragma unroll
    for (int i = 0; i < 2; ++i) {
      afr[i] = *(const v16bf*)(afp + i * 512);
      wfr[i] = *(const v16bf*)(wfp + i * 512);
    }
#pragma unroll
    for (int mi = 0; mi < 2; ++mi)
#pragma unroll
      for (int ni = 0; ni < 2; ++ni) {
        acc[mi][ni] = __builtin_amdgcn_wmma_f32_16x16x32_bf16(
            false, afr[mi], false, wfr[ni], (short)0, acc[mi][ni],
            false, false);
      }
    __syncthreads();
  }

  const int hi = lane >> 4;
#pragma unroll
  for (int mi = 0; mi < 2; ++mi) {
    const int rowb = tbase + (waveM * 2 + mi) * 16 + hi * 8;
#pragma unroll
    for (int ni = 0; ni < 2; ++ni) {
      const int col = nbase + (waveN * 2 + ni) * 16 + (lane & 15);
#pragma unroll
      for (int r = 0; r < 8; ++r) {
        out[(size_t)(rowb + r) * HD + col] = acc[mi][ni][r];
      }
    }
  }
}

// ---------------------------------------------------------------------------
extern "C" void kernel_launch(void* const* d_in, const int* in_sizes, int n_in,
                              void* d_out, int out_size, void* d_ws,
                              size_t ws_size, hipStream_t stream) {
  const float* x           = (const float*)d_in[0];
  const int*   gate_codes  = (const int*)d_in[1];
  const float* gate_absmax = (const float*)d_in[2];
  const int*   up_codes    = (const int*)d_in[3];
  const float* up_absmax   = (const float*)d_in[4];
  const int*   down_codes  = (const int*)d_in[5];
  const float* down_absmax = (const float*)d_in[6];
  float* out = (float*)d_out;

  // Workspace: H = silu(gate)*up as bf16, [TDIM, MDIM] = 58.7 MB
  unsigned short* H = (unsigned short*)d_ws;

  dim3 blk(256);
  dim3 g1(MDIM / 128, TDIM / 64);   // 112 x 32
  gateup_kernel<<<g1, blk, 0, stream>>>(x, gate_codes, gate_absmax,
                                        up_codes, up_absmax, H);
  dim3 g2(HD / 128, TDIM / 64);     // 32 x 32
  down_kernel<<<g2, blk, 0, stream>>>(H, down_codes, down_absmax, out);
}
